// Gen2SingleInputReadout_83614423318661
// MI455X (gfx1250) — matrix operand order repaired and checked by reading, not compiled, not run on hardware
//
#include <hip/hip_runtime.h>
#include <hip/hip_bf16.h>
#include <math.h>

// ---------------- problem constants (from setup_inputs) ----------------
#define TT 4096
#define BB 8
#define IN_DIM 512
#define DV 512
#define DK 128
#define CSZ 128
#define NCH (TT / CSZ)          // 32 chunks
#define MROWS (TT * BB)         // 32768 flattened (t,b) rows
#define EPSC 1e-8f

// padded leading dims: +8 bf16 (16B) keeps rows 16B-aligned -> ds_load_b128,
// and 272B row stride still permutes LDS banks.
#define LDP 136
#define LDPU (LDP / 2)          // 68 dwords per row
#define LDX 40
#define LDXU (LDX / 2)          // 20 dwords per row

typedef __attribute__((ext_vector_type(16))) __bf16       v16bf;
typedef __attribute__((ext_vector_type(8)))  float        v8f;
typedef __attribute__((ext_vector_type(4)))  unsigned int u32x4;
typedef __attribute__((ext_vector_type(8)))  unsigned int u32x8;

union BC16 { v16bf b; u32x4 q[2]; };

__device__ __forceinline__ v8f vzero8() {
  v8f z = {0.f, 0.f, 0.f, 0.f, 0.f, 0.f, 0.f, 0.f};
  return z;
}

__device__ __forceinline__ unsigned int pk_bf16(float lo, float hi) {
  union { __bf16 b[2]; unsigned int u; } t;
  t.b[0] = (__bf16)lo;
  t.b[1] = (__bf16)hi;
  return t.u;
}

// ---------------- packed WMMA operand loaders (wave32, 16x16x32 bf16) ----
// A: logical 16x32 (MxK), source row-major MxK, ld 16B-aligned -> 2x b128.
__device__ __forceinline__ v16bf load_A_pack(const __bf16* p, int ld) {
  int l = threadIdx.x & 31;
  int m = l & 15;
  int kh = (l >> 4) * 8;
  BC16 t;
  t.q[0] = *(const u32x4*)(p + m * ld + kh);        // K = kh+0..7
  t.q[1] = *(const u32x4*)(p + m * ld + kh + 16);   // K = 16+kh+0..7
  return t.b;
}

// B: logical 32x16 (KxN), source stored K-transposed (N x K row-major),
// i.e. B[k][n] = p[n*ld + k] -> 2x b128.
__device__ __forceinline__ v16bf load_B_pack(const __bf16* p, int ld) {
  int l = threadIdx.x & 31;
  int n = l & 15;
  int kh = (l >> 4) * 8;
  BC16 t;
  t.q[0] = *(const u32x4*)(p + n * ld + kh);        // K = kh+0..7
  t.q[1] = *(const u32x4*)(p + n * ld + kh + 16);   // K = 16+kh+0..7
  return t.b;
}

__device__ __forceinline__ v8f wmma_bf16(v16bf a, v16bf b, v8f c) {
  return __builtin_amdgcn_wmma_f32_16x16x32_bf16(
      false, a, false, b, (short)0, c, false, false);
}

// ---------------- Tensor Data Mover: 128x128 bf16 tile -> padded LDS ----
// D# group0/group1 per CDNA5 ISA ch.8.  data_size=2B; tile=tensor=128x128;
// row stride 1024 elements (B*DK); pad_enable with interval=64 dwords,
// amount=4 dwords -> LDS rows of 272B == LDP(136) bf16.  Tracked by
// TENSORcnt; one issue per (branch-executing) wave.
__device__ __forceinline__ void tdm_load_tile_128x128(unsigned lds_off,
                                                      const __bf16* gptr) {
  unsigned long long ga = (unsigned long long)gptr;
  u32x4 g0;
  g0[0] = 1u;                                        // count=1, user mode
  g0[1] = lds_off;                                   // lds_addr (bytes)
  g0[2] = (unsigned)(ga & 0xffffffffu);              // global_addr[31:0]
  g0[3] = (unsigned)((ga >> 32) & 0x1ffffffu)        // global_addr[56:32]
          | (2u << 30);                              // type=2 ("image")
  u32x8 g1;
  g1[0] = (1u << 16)        // data_size = 1 -> 2 bytes
        | (1u << 20)        // pad_enable
        | (5u << 22)        // pad_interval: 64 dwords stored ...
        | (3u << 25);       // pad_amount: ... then 4 dwords pad
  g1[1] = (128u & 0xffffu) << 16;                    // tensor_dim0[15:0]
  g1[2] = (128u >> 16) | ((128u & 0xffffu) << 16);   // dim0 hi | dim1 lo
  g1[3] = (128u >> 16) | (128u << 16);               // dim1 hi | tile_dim0
  g1[4] = 128u;                                      // tile_dim1 (tile_dim2=0)
  g1[5] = 1024u;                                     // tensor_dim0_stride lo32
  g1[6] = 0u;                                        // stride hi | dim1_stride lo
  g1[7] = 0u;
  asm volatile("tensor_load_to_lds %0, %1" :: "s"(g0), "s"(g1) : "memory");
}

// ======================================================================
// Kernel 1: fused projection GEMM  (M=32768, K=512, N=896 concat)
//   cols [0,512)->v (bf16), [512,640)->k, [640,768)->q, [768,896)->sigmoid=a
// Block = 256 thr (8 waves), tile 128(M) x 64(N), K-step 32 via LDS.
// ======================================================================
__global__ __launch_bounds__(256) void proj_kernel(
    const float* __restrict__ x,
    const float* __restrict__ Wv, const float* __restrict__ bv,
    const float* __restrict__ Wk, const float* __restrict__ bk,
    const float* __restrict__ Wq, const float* __restrict__ bq,
    const float* __restrict__ Wa, const float* __restrict__ ba,
    __bf16* __restrict__ Vb, float* __restrict__ Kb,
    float* __restrict__ Qb, float* __restrict__ Ab) {
  __shared__ __bf16 Xs[128 * LDX];  // 10 KB
  __shared__ __bf16 Ws[64 * LDX];   // 5 KB
  unsigned int* Xu = (unsigned int*)Xs;
  unsigned int* Wu = (unsigned int*)Ws;

  const int m0 = blockIdx.x * 128;
  const int n0 = blockIdx.y * 64;

  const float* Wm;
  const float* bias;
  int region, ncols, ncol0;
  if (n0 < DV)             { Wm = Wv; bias = bv; region = 0; ncols = DV; ncol0 = n0; }
  else if (n0 < DV + DK)   { Wm = Wk; bias = bk; region = 1; ncols = DK; ncol0 = n0 - DV; }
  else if (n0 < DV + 2*DK) { Wm = Wq; bias = bq; region = 2; ncols = DK; ncol0 = n0 - DV - DK; }
  else                     { Wm = Wa; bias = ba; region = 3; ncols = DK; ncol0 = n0 - DV - 2*DK; }

  const int tid   = threadIdx.x;
  const int wave  = tid >> 5;
  const int l     = tid & 31;
  const int colL  = l & 15;
  const int rbase = (l >> 4) * 8;

  v8f acc[4];
#pragma unroll
  for (int i = 0; i < 4; ++i) acc[i] = vzero8();

  for (int k0 = 0; k0 < IN_DIM; k0 += 32) {
    // X tile: 128 rows x 16 k-pairs; float2 global load -> packed u32 store
    for (int i = tid; i < 128 * 16; i += 256) {
      int r = i >> 4, kp = i & 15;
      const float2 v2 =
          *(const float2*)(x + (size_t)(m0 + r) * IN_DIM + k0 + 2 * kp);
      Xu[r * LDXU + kp] = pk_bf16(v2.x, v2.y);
    }
    // W tile transposed (N x K): two strided b32 reads, packed u32 store
    for (int i = tid; i < 64 * 16; i += 256) {
      int c = i & 63, kp = i >> 6;
      float w0 = Wm[(size_t)(k0 + 2 * kp) * ncols + ncol0 + c];
      float w1 = Wm[(size_t)(k0 + 2 * kp + 1) * ncols + ncol0 + c];
      Wu[c * LDXU + kp] = pk_bf16(w0, w1);
    }
    __syncthreads();
    v16bf a = load_A_pack(Xs + wave * 16 * LDX, LDX);
#pragma unroll
    for (int ct = 0; ct < 4; ++ct) {
      v16bf b = load_B_pack(Ws + ct * 16 * LDX, LDX);
      acc[ct] = wmma_bf16(a, b, acc[ct]);
    }
    __syncthreads();
  }

#pragma unroll
  for (int ct = 0; ct < 4; ++ct) {
#pragma unroll
    for (int r = 0; r < 8; ++r) {
      int gm = m0 + wave * 16 + rbase + r;
      int gc = ncol0 + ct * 16 + colL;
      float val = acc[ct][r] + bias[gc];
      if (region == 0)      Vb[(size_t)gm * DV + gc] = (__bf16)val;
      else if (region == 1) Kb[(size_t)gm * DK + gc] = val;
      else if (region == 2) Qb[(size_t)gm * DK + gc] = val;
      else                  Ab[(size_t)gm * DK + gc] = 1.0f / (1.0f + __expf(-val));
    }
  }
}

// ======================================================================
// Kernel 2: per-(chunk,b,n) cumprod gate, fp32 recurrence, bf16 outputs.
// ======================================================================
__global__ __launch_bounds__(256) void gate_kernel(
    const float* __restrict__ Kb, const float* __restrict__ Qb,
    const float* __restrict__ Ab,
    __bf16* __restrict__ QPb, __bf16* __restrict__ KPb,
    __bf16* __restrict__ WTb, float* __restrict__ PEnd) {
  int gid = blockIdx.x * blockDim.x + threadIdx.x;
  int n  = gid & 127;
  int b  = (gid >> 7) & 7;
  int ch = gid >> 10;
  if (ch >= NCH) return;

  const size_t base   = ((size_t)(ch * CSZ) * BB + b) * DK + n;
  const size_t stride = (size_t)BB * DK;

  float p = 1.f;
  for (int t = 0; t < CSZ; ++t) p *= fmaxf(Ab[base + t * stride], EPSC);
  const float pend = p;
  PEnd[((size_t)ch * BB + b) * DK + n] = pend;

  p = 1.f;
  for (int t = 0; t < CSZ; ++t) {
    size_t idx = base + t * stride;
    p *= fmaxf(Ab[idx], EPSC);
    float kp = Kb[idx] / (p + EPSC);
    QPb[idx] = (__bf16)(Qb[idx] * p);
    KPb[idx] = (__bf16)kp;
    WTb[idx] = (__bf16)(kp * pend);
  }
}

// ---------------- scan kernel LDS layout (bytes, all 16B aligned) -------
#define OFF_SST 0u          // fp32 S: 64*128*4      = 32768
#define OFF_SSB 32768u      // bf16 S: 64*136*2      = 17408
#define OFF_QP  50176u      // 128*136*2 = 34816, x2 buffers
#define OFF_KW  119808u     // kp then masked W, 34816 x2
#define OFF_WT  189440u     // wt^T (n x t), 34816 x2
#define OFF_VT  259072u     // V^T (d x t), 17408 x2
#define BUF_BIG 34816u
#define BUF_SML 17408u
#define SMEM_TOTAL 293888

// registers for software-pipelined staging of wt^T / V^T
struct StageRegs {
  unsigned wa[16], wb[16];
  unsigned va[8],  vb[8];
};

__device__ __forceinline__ void load_wtv_regs(const unsigned* WTu,
                                              const unsigned* Vu, int tg0,
                                              int bb, int d0, int tid,
                                              StageRegs& r) {
#pragma unroll
  for (int it = 0; it < 16; ++it) {
    int i = tid + it * 256;
    int np = i & 63, tp = i >> 6;
    r.wa[it] = WTu[((size_t)(tg0 + 2 * tp)     * BB + bb) * 64 + np];
    r.wb[it] = WTu[((size_t)(tg0 + 2 * tp + 1) * BB + bb) * 64 + np];
  }
#pragma unroll
  for (int it = 0; it < 8; ++it) {
    int i = tid + it * 256;
    int dp = i & 31, tp = i >> 5;
    r.va[it] = Vu[((size_t)(tg0 + 2 * tp)     * BB + bb) * (DV / 2) + (d0 >> 1) + dp];
    r.vb[it] = Vu[((size_t)(tg0 + 2 * tp + 1) * BB + bb) * (DV / 2) + (d0 >> 1) + dp];
  }
}

__device__ __forceinline__ void store_wtv_lds(unsigned* uWT, unsigned* uVt,
                                              int tid, const StageRegs& r) {
#pragma unroll
  for (int it = 0; it < 16; ++it) {
    int i = tid + it * 256;
    int np = i & 63, tp = i >> 6;
    unsigned ua = r.wa[it], ub = r.wb[it];
    uWT[(2 * np)     * LDPU + tp] = (ua & 0xffffu) | (ub << 16);
    uWT[(2 * np + 1) * LDPU + tp] = (ua >> 16) | (ub & 0xffff0000u);
  }
#pragma unroll
  for (int it = 0; it < 8; ++it) {
    int i = tid + it * 256;
    int dp = i & 31, tp = i >> 5;
    unsigned ua = r.va[it], ub = r.vb[it];
    uVt[(2 * dp)     * LDPU + tp] = (ua & 0xffffu) | (ub << 16);
    uVt[(2 * dp + 1) * LDPU + tp] = (ua >> 16) | (ub & 0xffff0000u);
  }
}

// ======================================================================
// Kernel 3: sequential chunk scan. Grid (batch=8, dblock=8 of 64 d-cols).
// Double-buffered: qp/kp tiles prefetched by the Tensor Data Mover (pad
// injected by the D# to produce the 16B-padded LDS layout directly);
// wt^T/V^T software-pipelined through registers.  fp32 S + bf16 shadow
// persist in LDS across chunks.
// NOTE: LDS buffer pointers are always derived from smem + integer offset
// inside the loop (no pointer arrays -> no addrspacecast initializers).
// ======================================================================
__global__ __launch_bounds__(256) void scan_kernel(
    const __bf16* __restrict__ Vb, const __bf16* __restrict__ QPb,
    const __bf16* __restrict__ KPb, const __bf16* __restrict__ WTb,
    const float* __restrict__ PEnd, float* __restrict__ out) {
  extern __shared__ char smem[];
  float*  Sst = (float*)(smem + OFF_SST);
  __bf16* sSb = (__bf16*)(smem + OFF_SSB);

  const int bb = blockIdx.x;       // batch
  const int d0 = blockIdx.y * 64;  // d block start
  const int tid   = threadIdx.x;
  const int wave  = tid >> 5;
  const int l     = tid & 31;
  const int colL  = l & 15;
  const int rbase = (l >> 4) * 8;

  const unsigned* WTu = (const unsigned*)WTb;
  const unsigned* Vu  = (const unsigned*)Vb;

  for (int i = tid; i < 64 * 128; i += 256) Sst[i] = 0.f;
  for (int i = tid; i < 64 * LDP; i += 256) sSb[i] = (__bf16)0.f;

  // ---- prologue: prefetch chunk 0 into buffer 0
  if (wave == 0) {
    tdm_load_tile_128x128(OFF_QP, QPb + (size_t)bb * DK);
    tdm_load_tile_128x128(OFF_KW, KPb + (size_t)bb * DK);
  }
  {
    StageRegs sr;
    load_wtv_regs(WTu, Vu, 0, bb, d0, tid, sr);
    store_wtv_lds((unsigned*)(smem + OFF_WT), (unsigned*)(smem + OFF_VT), tid, sr);
  }
  __builtin_amdgcn_s_wait_tensorcnt(0);
  __syncthreads();

  for (int ch = 0; ch < NCH; ++ch) {
    const int tg0 = ch * CSZ;
    const unsigned cur = (unsigned)(ch & 1);
    const unsigned nxt = cur ^ 1u;
    const bool pf = (ch + 1 < NCH);

    __bf16*   sQPc = (__bf16*)(smem + OFF_QP + cur * BUF_BIG);
    __bf16*   sKWc = (__bf16*)(smem + OFF_KW + cur * BUF_BIG);
    __bf16*   sVtc = (__bf16*)(smem + OFF_VT + cur * BUF_SML);
    __bf16*   sWTc = (__bf16*)(smem + OFF_WT + cur * BUF_BIG);

    // ---- issue TDM prefetch of next chunk's qp/kp (async, TENSORcnt)
    if (pf && wave == 0) {
      size_t gn = (size_t)((ch + 1) * CSZ) * BB + bb;
      tdm_load_tile_128x128(OFF_QP + nxt * BUF_BIG, QPb + gn * DK);
      tdm_load_tile_128x128(OFF_KW + nxt * BUF_BIG, KPb + gn * DK);
    }

    // ---- step A: W = tril(qp @ kp^T); wave w owns rows [16w,16w+16)
    v8f wacc[8];
#pragma unroll
    for (int ct = 0; ct < 8; ++ct) wacc[ct] = vzero8();
    for (int k0 = 0; k0 < DK; k0 += 32) {
      v16bf a = load_A_pack(sQPc + wave * 16 * LDP + k0, LDP);
#pragma unroll
      for (int ct = 0; ct < 8; ++ct) {
        v16bf b = load_B_pack(sKWc + ct * 16 * LDP + k0, LDP);  // kp = N x K
        wacc[ct] = wmma_bf16(a, b, wacc[ct]);
      }
    }
    __syncthreads();  // all kp reads done before overwriting with W
#pragma unroll
    for (int ct = 0; ct < 8; ++ct) {
#pragma unroll
      for (int r = 0; r < 8; ++r) {
        int t = wave * 16 + rbase + r;
        int s = ct * 16 + colL;
        sKWc[t * LDP + s] = (__bf16)((t >= s) ? wacc[ct][r] : 0.f);
      }
    }
    __syncthreads();

    // ---- step B: y = W @ V + qp @ S^T
#pragma unroll
    for (int ct = 0; ct < 4; ++ct) {
      v8f acc = vzero8();
      for (int k0 = 0; k0 < CSZ; k0 += 32) {             // intra-chunk W @ V
        v16bf a = load_A_pack(sKWc + wave * 16 * LDP + k0, LDP);
        v16bf b = load_B_pack(sVtc + ct * 16 * LDP + k0, LDP);
        acc = wmma_bf16(a, b, acc);
      }
      for (int k0 = 0; k0 < DK; k0 += 32) {              // carry: qp @ S^T
        v16bf a = load_A_pack(sQPc + wave * 16 * LDP + k0, LDP);
        v16bf b = load_B_pack(sSb + ct * 16 * LDP + k0, LDP);
        acc = wmma_bf16(a, b, acc);
      }
#pragma unroll
      for (int r = 0; r < 8; ++r) {
        int t = wave * 16 + rbase + r;
        int c = ct * 16 + colL;
        out[((size_t)(tg0 + t) * BB + bb) * DV + d0 + c] = acc[r];
      }
    }

    // ---- software-pipelined global loads of next chunk's wt/V into regs
    StageRegs sr;
    if (pf) load_wtv_regs(WTu, Vu, tg0 + CSZ, bb, d0, tid, sr);

    __syncthreads();  // all reads of old S (step B) done before update

    // ---- step C: S = S * p_end + V^T @ wt
    const int rs = wave >> 1;
    const int ctbase = (wave & 1) * 4;
#pragma unroll
    for (int cti = 0; cti < 4; ++cti) {
      int ct = ctbase + cti;
      v8f acc = vzero8();
      for (int k0 = 0; k0 < CSZ; k0 += 32) {
        v16bf a = load_A_pack(sVtc + rs * 16 * LDP + k0, LDP);
        v16bf b = load_B_pack(sWTc + ct * 16 * LDP + k0, LDP);
        acc = wmma_bf16(a, b, acc);
      }
#pragma unroll
      for (int r = 0; r < 8; ++r) {
        int dc = rs * 16 + rbase + r;
        int nn = ct * 16 + colL;
        float pe = PEnd[((size_t)ch * BB + bb) * DK + nn];
        float sv = Sst[dc * DK + nn] * pe + acc[r];
        Sst[dc * DK + nn] = sv;
        sSb[dc * LDP + nn] = (__bf16)sv;   // bf16 shadow for next chunk
      }
    }

    // ---- drain pipeline: store staged wt/V, wait TDM, hand off buffers
    if (pf) {
      store_wtv_lds((unsigned*)(smem + OFF_WT + nxt * BUF_BIG),
                    (unsigned*)(smem + OFF_VT + nxt * BUF_SML), tid, sr);
    }
    __builtin_amdgcn_s_wait_tensorcnt(0);
    __syncthreads();
  }
}

// ======================================================================
extern "C" void kernel_launch(void* const* d_in, const int* in_sizes, int n_in,
                              void* d_out, int out_size, void* d_ws, size_t ws_size,
                              hipStream_t stream) {
  const float* x  = (const float*)d_in[0];
  const float* Wv = (const float*)d_in[1];
  const float* bv = (const float*)d_in[2];
  const float* Wk = (const float*)d_in[3];
  const float* bk = (const float*)d_in[4];
  const float* Wq = (const float*)d_in[5];
  const float* bq = (const float*)d_in[6];
  const float* Wa = (const float*)d_in[7];
  const float* ba = (const float*)d_in[8];
  float* out = (float*)d_out;

  // workspace layout (bytes)
  char* ws = (char*)d_ws;
  __bf16* Vb  = (__bf16*)(ws);                               // 32 MB
  char* p = ws + (size_t)MROWS * DV * 2;
  float* Kb   = (float*)p;  p += (size_t)MROWS * DK * 4;     // 16 MB
  float* Qb   = (float*)p;  p += (size_t)MROWS * DK * 4;     // 16 MB
  float* Ab   = (float*)p;  p += (size_t)MROWS * DK * 4;     // 16 MB
  __bf16* QPb = (__bf16*)p; p += (size_t)MROWS * DK * 2;     // 8 MB
  __bf16* KPb = (__bf16*)p; p += (size_t)MROWS * DK * 2;     // 8 MB
  __bf16* WTb = (__bf16*)p; p += (size_t)MROWS * DK * 2;     // 8 MB
  float* PEnd = (float*)p;                                   // 128 KB

  (void)hipFuncSetAttribute((const void*)scan_kernel,
                            hipFuncAttributeMaxDynamicSharedMemorySize,
                            SMEM_TOTAL);

  // 1) fused projections: grid 256 x 14 tiles of 128x64
  proj_kernel<<<dim3(MROWS / 128, (DV + 3 * DK) / 64), 256, 0, stream>>>(
      x, Wv, bv, Wk, bk, Wq, bq, Wa, ba, Vb, Kb, Qb, Ab);

  // 2) gates/cumprod: 32*8*128 = 32768 threads
  gate_kernel<<<NCH * BB * DK / 256, 256, 0, stream>>>(
      Kb, Qb, Ab, QPb, KPb, WTb, PEnd);

  // 3) chunk scan: 8 batches x 8 d-blocks, 287 KB dynamic LDS each
  scan_kernel<<<dim3(BB, DV / 64), 256, SMEM_TOTAL, stream>>>(
      Vb, QPb, KPb, WTb, PEnd, out);
}
